// MaskRCNN_38027640439385
// MI455X (gfx1250) — matrix operand order repaired and checked by reading, not compiled, not run on hardware
//
#include <hip/hip_runtime.h>
#include <hip/hip_bf16.h>

typedef __attribute__((ext_vector_type(16))) _Float16 v16h;
typedef __attribute__((ext_vector_type(8)))  float    v8f;

#define BATCH_N 32
#define GRID_C  14
#define NCELL   196

// ---------------------------------------------------------------------------
// elementwise f32 -> f16
__global__ void f32_to_f16_k(const float* __restrict__ src, _Float16* __restrict__ dst, int n) {
  int i = blockIdx.x * blockDim.x + threadIdx.x;
  if (i < n) dst[i] = (_Float16)src[i];
}

// ---------------------------------------------------------------------------
// pack OIHW f32 weights into f16 [Kpad][Npad], k = (ci*KH + r)*KW + s, col = cout
__global__ void pack_w_k(const float* __restrict__ src, _Float16* __restrict__ dst,
                         int K, int Kpad, int Cout, int Npad) {
  int i = blockIdx.x * blockDim.x + threadIdx.x;
  int tot = Kpad * Npad;
  if (i >= tot) return;
  int k = i / Npad, n = i - k * Npad;
  float v = 0.f;
  if (k < K && n < Cout) v = src[(size_t)n * K + k];
  dst[i] = (_Float16)v;
}

// ---------------------------------------------------------------------------
// implicit-GEMM conv via v_wmma_f32_16x16x32_f16.
// Block: 256 threads = 8 waves. Wave w -> M-tile (blockIdx.x*8+w).
// blockIdx.y -> group of NT N-tiles: each wave computes a 16M x (16*NT)N tile so the
// expensive per-lane A im2col gather is amortized over NT WMMAs.
// B panel (32K x 16N per sub-tile) staged in LDS, swizzled so each lane reads its
// v16h fragment as one contiguous 32-byte ds_load.
template<int KH, int KW, int NT>
__global__ __launch_bounds__(256) void conv_wmma_k(
    const _Float16* __restrict__ in, const _Float16* __restrict__ wpk,
    const float* __restrict__ scale, const float* __restrict__ bias,
    const _Float16* __restrict__ residual,
    _Float16* __restrict__ outh, float* __restrict__ outf,
    int Cin, int Cout, int Npad, int Hin, int Win, int Ho, int Wo,
    int stride, int pad, int relu, int mode, int nvalid)
{
  __shared__ _Float16 ldsB[512 * NT];
  const int lane = threadIdx.x & 31;
  const int wave = threadIdx.x >> 5;
  const int half = lane >> 4;
  const int l16  = lane & 15;
  const int K      = Cin * KH * KW;
  const int ksteps = (K + 31) >> 5;
  const int ntile0 = blockIdx.y * NT;
  const int HoWo   = Ho * Wo;
  const int m0     = (blockIdx.x * 8 + wave) * 16;

  // A-row for this lane: M = m0 + l16  (A layout: lanes 0-15 and 16-31 both hold M=0..15)
  const int m    = m0 + l16;
  const int bimg = m / HoWo;
  const int cell = m - bimg * HoWo;
  const int oy   = cell / Wo;
  const int ox   = cell - oy * Wo;
  const int iy0  = oy * stride - pad;
  const int ix0  = ox * stride - pad;
  const _Float16* inb  = in + (size_t)bimg * Cin * Hin * Win;
  const _Float16* wcol = wpk + (size_t)ntile0 * 16;

  v8f acc[NT] = {};
  for (int ks = 0; ks < ksteps; ++ks) {
    const int kbase = ks << 5;
    // cooperative swizzled stage of B panels: value B[kbase+kk][nt*16+n] stored so
    // consumer lane l = ((kk>>3)&1)*16 + n reads element ((kk>>4)<<3)|(kk&7) of sub-panel nt contiguously.
    for (int e = threadIdx.x; e < 512 * NT; e += 256) {
      int nt = e >> 9;
      int w9 = e & 511;
      int kk = w9 >> 4, n = w9 & 15;
      _Float16 v = wcol[(size_t)(kbase + kk) * Npad + nt * 16 + n];
      ldsB[(nt << 9) | (((((kk >> 3) & 1) << 4) | n) << 4) | (((kk >> 4) << 3) | (kk & 7))] = v;
    }
    // prefetch next K-step's weight rows (lowers to global_prefetch_b8)
    if (ks + 1 < ksteps && threadIdx.x < 32)
      __builtin_prefetch(wcol + (size_t)(kbase + 32 + threadIdx.x) * Npad, 0, 1);
    __syncthreads();

    // A fragment gather (16-bit A 16x32 layout): VGPR v pair-base K:
    //   v<4 : 2v + 8*half ;  v>=4 : 16 + 2(v-4) + 8*half
    v16h afrag;
#pragma unroll
    for (int v = 0; v < 8; ++v) {
      const int kb = ((v < 4) ? (v << 1) : (16 + ((v - 4) << 1))) + (half << 3);
#pragma unroll
      for (int p = 0; p < 2; ++p) {
        const int k = kbase + kb + p;
        _Float16 val = (_Float16)0.f;
        if (k < K) {
          const int ci = k / (KH * KW);
          const int rm = k - ci * (KH * KW);
          const int rr = rm / KW;
          const int ss = rm - rr * KW;
          const int iy = iy0 + rr, ix = ix0 + ss;
          if ((unsigned)iy < (unsigned)Hin && (unsigned)ix < (unsigned)Win)
            val = inb[((size_t)ci * Hin + iy) * Win + ix];
        }
        afrag[2 * v + p] = val;
      }
    }

#pragma unroll
    for (int nt = 0; nt < NT; ++nt) {
      v16h bfrag = *(const v16h*)(ldsB + (nt << 9) + lane * 16);
      acc[nt] = __builtin_amdgcn_wmma_f32_16x16x32_f16(false, afrag, false, bfrag,
                                                       (short)0, acc[nt], false, false);
    }
    __syncthreads();
  }

  // epilogue: D layout lane l holds N=l&15, VGPR r -> M = r + 8*(l>>4)
  float scA[NT], biA[NT];
#pragma unroll
  for (int nt = 0; nt < NT; ++nt) {
    const int co = (ntile0 + nt) * 16 + l16;
    scA[nt] = 1.f; biA[nt] = 0.f;
    if (co < Cout) {
      if (scale) scA[nt] = scale[co];
      if (bias)  biA[nt] = bias[co];
    }
  }
#pragma unroll
  for (int r = 0; r < 8; ++r) {
    const int mrow = m0 + (half << 3) + r;
    if (mode == 0) {
      const int b2 = mrow / HoWo;
      const int c2 = mrow - b2 * HoWo;
      const size_t base = (size_t)b2 * Cout * HoWo + c2;
#pragma unroll
      for (int nt = 0; nt < NT; ++nt) {
        const int co = (ntile0 + nt) * 16 + l16;
        float y = acc[nt][r] * scA[nt] + biA[nt];
        const size_t oidx = base + (size_t)co * HoWo;
        if (residual) y += (float)residual[oidx];
        if (relu && y < 0.f) y = 0.f;
        outh[oidx] = (_Float16)y;
      }
    } else {
      const int co = ntile0 * 16 + l16;
      if (co < nvalid)
        outf[(size_t)mrow * nvalid + co] = acc[0][r] * scA[0] + biA[0];
    }
  }
}

// ---------------------------------------------------------------------------
// 3x3 stride-2 pad-1 maxpool, f16
__global__ void maxpool_k(const _Float16* __restrict__ in, _Float16* __restrict__ out,
                          int BC, int Hin, int Win, int Ho, int Wo) {
  int idx = blockIdx.x * blockDim.x + threadIdx.x;
  int tot = BC * Ho * Wo;
  if (idx >= tot) return;
  int bc  = idx / (Ho * Wo);
  int rem = idx - bc * (Ho * Wo);
  int y = rem / Wo, x = rem - y * Wo;
  const _Float16* p = in + (size_t)bc * Hin * Win;
  float mx = -3.0e38f;
  for (int r = 0; r < 3; ++r) {
    int iy = y * 2 - 1 + r;
    if ((unsigned)iy >= (unsigned)Hin) continue;
    for (int s = 0; s < 3; ++s) {
      int ix = x * 2 - 1 + s;
      if ((unsigned)ix >= (unsigned)Win) continue;
      float v = (float)p[(size_t)iy * Win + ix];
      mx = fmaxf(mx, v);
    }
  }
  out[idx] = (_Float16)mx;
}

// ---------------------------------------------------------------------------
// anchors + softmax scores + box decode (one thread per (b, cell))
__global__ void decode_k(const float* __restrict__ cls, const float* __restrict__ reg,
                         float* __restrict__ scores, float* __restrict__ boxes,
                         float* __restrict__ anchorsOut) {
  int idx = blockIdx.x * blockDim.x + threadIdx.x;
  if (idx >= BATCH_N * NCELL) return;
  int cell = idx % NCELL;
  int oy = cell / GRID_C, ox = cell - oy * GRID_C;
  float cxa = (ox + 0.5f) * 16.f, cya = (oy + 0.5f) * 16.f;
  if (idx < NCELL) {  // b == 0 writes anchors
    anchorsOut[cell * 4 + 0] = cxa - 16.f;
    anchorsOut[cell * 4 + 1] = cya - 16.f;
    anchorsOut[cell * 4 + 2] = cxa + 16.f;
    anchorsOut[cell * 4 + 3] = cya + 16.f;
  }
  float l0 = cls[idx * 2 + 0], l1 = cls[idx * 2 + 1];
  float mx = fmaxf(l0, l1);
  float e0 = __expf(l0 - mx), e1 = __expf(l1 - mx);
  scores[idx] = e1 / (e0 + e1);
  float d0 = reg[idx * 4 + 0], d1 = reg[idx * 4 + 1];
  float d2 = reg[idx * 4 + 2], d3 = reg[idx * 4 + 3];
  float cx = cxa + d0 * 32.f, cy = cya + d1 * 32.f;
  float w = 32.f * __expf(d2), h = 32.f * __expf(d3);
  boxes[idx * 4 + 0] = cx - 0.5f * w;
  boxes[idx * 4 + 1] = cy - 0.5f * h;
  boxes[idx * 4 + 2] = cx + 0.5f * w;
  boxes[idx * 4 + 3] = cy + 0.5f * h;
}

// ---------------------------------------------------------------------------
// greedy NMS, one lane per image, replicating the reference fori_loop semantics
__global__ void nms_k(const float* __restrict__ scores, const float* __restrict__ boxes,
                      float* __restrict__ stmp, int* __restrict__ order, int* __restrict__ supp,
                      float* __restrict__ maskOut) {
  int b = blockIdx.x * blockDim.x + threadIdx.x;
  if (b >= BATCH_N) return;
  const float* sc = scores + b * NCELL;
  const float* bx = boxes  + (size_t)b * NCELL * 4;
  float* st = stmp  + b * NCELL;
  int*   od = order + b * NCELL;
  int*   sp = supp  + b * NCELL;
  for (int i = 0; i < NCELL; ++i) { st[i] = sc[i]; sp[i] = 0; }
  // stable argsort(-scores) via selection (strict > keeps earliest index on ties)
  for (int i = 0; i < NCELL; ++i) {
    int best = 0; float bv = -1e30f;
    for (int j = 0; j < NCELL; ++j) { float v = st[j]; if (v > bv) { bv = v; best = j; } }
    od[i] = best; st[best] = -1e30f;
  }
  int suppCount = 0;
  for (int i = 0; i < NCELL; ++i) {
    int oi = od[i];
    int remaining = NCELL - suppCount;
    int do_keep = (!sp[i]) && (remaining > 1);
    if (do_keep) {
      float ax1 = bx[oi*4+0], ay1 = bx[oi*4+1], ax2 = bx[oi*4+2], ay2 = bx[oi*4+3];
      float aarea = (ax2 - ax1) * (ay2 - ay1);
      for (int j = 0; j < NCELL; ++j) {
        if (sp[j]) continue;
        int oj = od[j];
        float bx1 = bx[oj*4+0], by1 = bx[oj*4+1], bx2 = bx[oj*4+2], by2 = bx[oj*4+3];
        float barea = (bx2 - bx1) * (by2 - by1);
        float ix1 = fmaxf(ax1, bx1), iy1 = fmaxf(ay1, by1);
        float ix2 = fminf(ax2, bx2), iy2 = fminf(ay2, by2);
        float iw = fmaxf(ix2 - ix1, 0.f), ih = fmaxf(iy2 - iy1, 0.f);
        float inter = iw * ih;
        float iou = inter / (aarea + barea - inter + 1e-9f);
        if (iou > 0.3f) { sp[j] = 1; ++suppCount; }   // includes self
      }
    }
    maskOut[b * NCELL + oi] = (do_keep && (sc[oi] > 0.5f)) ? 1.0f : 0.0f;
  }
}

// ---------------------------------------------------------------------------
extern "C" void kernel_launch(void* const* d_in, const int* in_sizes, int n_in,
                              void* d_out, int out_size, void* d_ws, size_t ws_size,
                              hipStream_t stream) {
  (void)n_in; (void)out_size; (void)ws_size;
  char* ws = (char*)d_ws;
  size_t off = 0;
  auto alloc = [&](size_t bytes) { size_t r = off; off += (bytes + 255) & ~(size_t)255; return r; };

  // Input index maps: order A = dict insertion order (x first), order B = jax pytree
  // alphabetical flatten (params sorted, x last). Detect via in_sizes[0].
  const bool insOrder = (in_sizes[0] == BATCH_N * 3 * 224 * 224);
  // conv list order: conv1, l1b0c1, l1b0c2, l1b1c1, l1b1c2, l2b0c1, l2b0c2, l2b0dn,
  //                  l2b1c1, l2b1c2, l3b0c1, l3b0c2, l3b0dn, l3b1c1, l3b1c2,
  //                  rpn1, rpn2, cls, reg
  static const int wA[19] = {1,4,7,10,13,16,19,22,25,28,31,34,37,40,43,46,49,52,54};
  static const int sA[19] = {2,5,8,11,14,17,20,23,26,29,32,35,38,41,44,47,50,53,55};
  static const int wB[19] = {4,9,10,15,16,21,22,25,30,31,36,37,40,45,46,53,54,3,48};
  static const int sB[19] = {0,5,7,11,13,17,19,23,26,28,32,34,38,41,43,49,51,2,47};
  const int* WI = insOrder ? wA : wB;
  const int* SI = insOrder ? sA : sB;
  const int  XI = insOrder ? 0 : 55;

  struct CW { int Cin, Cout, KH, KW; size_t pk; int Kpad, Npad; };
  CW cw[19] = {
    {3,64,7,7},   {64,64,3,3},  {64,64,3,3},  {64,64,3,3},  {64,64,3,3},
    {64,128,3,3}, {128,128,3,3},{64,128,1,1}, {128,128,3,3},{128,128,3,3},
    {128,256,3,3},{256,256,3,3},{128,256,1,1},{256,256,3,3},{256,256,3,3},
    {256,512,3,3},{512,256,3,3},{256,2,1,1},  {256,4,1,1},
  };
  for (int i = 0; i < 19; ++i) {
    int K = cw[i].Cin * cw[i].KH * cw[i].KW;
    cw[i].Kpad = (K + 31) & ~31;
    cw[i].Npad = (cw[i].Cout + 15) & ~15;
    cw[i].pk = alloc((size_t)cw[i].Kpad * cw[i].Npad * 2);
  }

  size_t oXH = alloc((size_t)BATCH_N * 3 * 224 * 224 * 2);
  size_t oA1 = alloc((size_t)BATCH_N * 64 * 112 * 112 * 2);
  size_t oP1 = alloc((size_t)BATCH_N * 64 * 56 * 56 * 2);
  size_t oBX = alloc((size_t)6422528 * 2);
  size_t oBY = alloc((size_t)6422528 * 2);
  size_t oBS = alloc((size_t)3309568 * 2);
  size_t oBT = alloc((size_t)3309568 * 2);
  size_t oSc = alloc((size_t)BATCH_N * NCELL * 4);
  size_t oBx = alloc((size_t)BATCH_N * NCELL * 16);
  size_t oSt = alloc((size_t)BATCH_N * NCELL * 4);
  size_t oOd = alloc((size_t)BATCH_N * NCELL * 4);
  size_t oSp = alloc((size_t)BATCH_N * NCELL * 4);

  auto F = [&](int i) { return (const float*)d_in[i]; };

  // pack all weights
  for (int i = 0; i < 19; ++i) {
    int K = cw[i].Cin * cw[i].KH * cw[i].KW;
    int tot = cw[i].Kpad * cw[i].Npad;
    pack_w_k<<<(tot + 255) / 256, 256, 0, stream>>>(
        F(WI[i]), (_Float16*)(ws + cw[i].pk), K, cw[i].Kpad, cw[i].Cout, cw[i].Npad);
  }

  auto conv = [&](int id, const void* I, void* O, const float* sc, const float* bi,
                  const void* R, int Hin, int Win, int Ho, int Wo,
                  int stride, int pad, int relu, int mode, int nvalid, float* outf) {
    const CW& c = cw[id];
    const _Float16* Wp = (const _Float16*)(ws + c.pk);
    const _Float16* Ip = (const _Float16*)I;
    const _Float16* Rp = (const _Float16*)R;
    _Float16* Op = (_Float16*)O;
    const int NT = (c.Npad % 64 == 0) ? 4 : 1;
    dim3 g((unsigned)(BATCH_N * Ho * Wo / 128), (unsigned)(c.Npad / (16 * NT)), 1);
    dim3 blk(256, 1, 1);
    if (c.KH == 7)
      conv_wmma_k<7,7,4><<<g, blk, 0, stream>>>(Ip, Wp, sc, bi, Rp, Op, outf,
          c.Cin, c.Cout, c.Npad, Hin, Win, Ho, Wo, stride, pad, relu, mode, nvalid);
    else if (c.KH == 3)
      conv_wmma_k<3,3,4><<<g, blk, 0, stream>>>(Ip, Wp, sc, bi, Rp, Op, outf,
          c.Cin, c.Cout, c.Npad, Hin, Win, Ho, Wo, stride, pad, relu, mode, nvalid);
    else if (NT == 4)
      conv_wmma_k<1,1,4><<<g, blk, 0, stream>>>(Ip, Wp, sc, bi, Rp, Op, outf,
          c.Cin, c.Cout, c.Npad, Hin, Win, Ho, Wo, stride, pad, relu, mode, nvalid);
    else
      conv_wmma_k<1,1,1><<<g, blk, 0, stream>>>(Ip, Wp, sc, bi, Rp, Op, outf,
          c.Cin, c.Cout, c.Npad, Hin, Win, Ho, Wo, stride, pad, relu, mode, nvalid);
  };

  // x -> f16
  int nx = BATCH_N * 3 * 224 * 224;
  f32_to_f16_k<<<(nx + 255) / 256, 256, 0, stream>>>(F(XI), (_Float16*)(ws + oXH), nx);

  // conv1 + bn1 + relu  (224 -> 112)
  conv(0, ws + oXH, ws + oA1, F(SI[0]), F(SI[0] + 1), nullptr, 224, 224, 112, 112, 2, 3, 1, 0, 0, nullptr);
  // maxpool (112 -> 56)
  int mp = BATCH_N * 64 * 56 * 56;
  maxpool_k<<<(mp + 255) / 256, 256, 0, stream>>>((const _Float16*)(ws + oA1),
                                                  (_Float16*)(ws + oP1), BATCH_N * 64, 112, 112, 56, 56);
  // layer1 block0
  conv(1, ws + oP1, ws + oBX, F(SI[1]), F(SI[1] + 1), nullptr, 56, 56, 56, 56, 1, 1, 1, 0, 0, nullptr);
  conv(2, ws + oBX, ws + oBY, F(SI[2]), F(SI[2] + 1), ws + oP1, 56, 56, 56, 56, 1, 1, 1, 0, 0, nullptr);
  // layer1 block1 (in-place residual)
  conv(3, ws + oBY, ws + oBX, F(SI[3]), F(SI[3] + 1), nullptr, 56, 56, 56, 56, 1, 1, 1, 0, 0, nullptr);
  conv(4, ws + oBX, ws + oBY, F(SI[4]), F(SI[4] + 1), ws + oBY, 56, 56, 56, 56, 1, 1, 1, 0, 0, nullptr);
  // layer2 block0 (downsample; 56 -> 28)
  conv(7, ws + oBY, ws + oBS, F(SI[7]), F(SI[7] + 1), nullptr, 56, 56, 28, 28, 2, 0, 0, 0, 0, nullptr);
  conv(5, ws + oBY, ws + oBX, F(SI[5]), F(SI[5] + 1), nullptr, 56, 56, 28, 28, 2, 1, 1, 0, 0, nullptr);
  conv(6, ws + oBX, ws + oBS, F(SI[6]), F(SI[6] + 1), ws + oBS, 28, 28, 28, 28, 1, 1, 1, 0, 0, nullptr);
  // layer2 block1
  conv(8, ws + oBS, ws + oBX, F(SI[8]), F(SI[8] + 1), nullptr, 28, 28, 28, 28, 1, 1, 1, 0, 0, nullptr);
  conv(9, ws + oBX, ws + oBS, F(SI[9]), F(SI[9] + 1), ws + oBS, 28, 28, 28, 28, 1, 1, 1, 0, 0, nullptr);
  // layer3 block0 (downsample; 28 -> 14)
  conv(12, ws + oBS, ws + oBY, F(SI[12]), F(SI[12] + 1), nullptr, 28, 28, 14, 14, 2, 0, 0, 0, 0, nullptr);
  conv(10, ws + oBS, ws + oBX, F(SI[10]), F(SI[10] + 1), nullptr, 28, 28, 14, 14, 2, 1, 1, 0, 0, nullptr);
  conv(11, ws + oBX, ws + oBY, F(SI[11]), F(SI[11] + 1), ws + oBY, 14, 14, 14, 14, 1, 1, 1, 0, 0, nullptr);
  // layer3 block1
  conv(13, ws + oBY, ws + oBX, F(SI[13]), F(SI[13] + 1), nullptr, 14, 14, 14, 14, 1, 1, 1, 0, 0, nullptr);
  conv(14, ws + oBX, ws + oBY, F(SI[14]), F(SI[14] + 1), ws + oBY, 14, 14, 14, 14, 1, 1, 1, 0, 0, nullptr);
  // RPN convs
  conv(15, ws + oBY, ws + oBT, F(SI[15]), F(SI[15] + 1), nullptr, 14, 14, 14, 14, 1, 1, 1, 0, 0, nullptr);
  conv(16, ws + oBT, ws + oBX, F(SI[16]), F(SI[16] + 1), nullptr, 14, 14, 14, 14, 1, 1, 1, 0, 0, nullptr);
  // heads -> d_out (f32, [B,196,C] layout)
  float* outp = (float*)d_out;
  conv(17, ws + oBX, nullptr, nullptr, F(SI[17]), nullptr, 14, 14, 14, 14, 1, 0, 0, 1, 2, outp);
  conv(18, ws + oBX, nullptr, nullptr, F(SI[18]), nullptr, 14, 14, 14, 14, 1, 0, 0, 1, 4, outp + 12544);

  // decode + anchors
  int nd = BATCH_N * NCELL;
  decode_k<<<(nd + 255) / 256, 256, 0, stream>>>(outp, outp + 12544,
      (float*)(ws + oSc), (float*)(ws + oBx), outp + 12544 + 25088);
  // per-image NMS -> keep mask (floats)
  nms_k<<<1, 32, 0, stream>>>((const float*)(ws + oSc), (const float*)(ws + oBx),
      (float*)(ws + oSt), (int*)(ws + oOd), (int*)(ws + oSp),
      outp + 12544 + 25088 + 784);
}
